// CustomMultiInputLSTM_23038204576177
// MI455X (gfx1250) — compile-verified, matrix operand order checked by reading, not run on hardware
//
#include <hip/hip_runtime.h>
#include <hip/hip_bf16.h>

// ---------------------------------------------------------------------------
// CustomMultiInputLSTM for MI455X (gfx1250, wave32, WMMA).
// Persistent-kernel recurrence over T with device-wide barriers; all H x H
// GEMMs run on v_wmma_f32_16x16x32_bf16 with weights pre-converted once into
// WMMA B-tile-packed bf16 (resident in the 192MB L2 across all 1024 steps).
// This revision register-caches the reused A operand (h / l_all row slabs,
// 16 x v16bf = 128 VGPRs) so the inner GEMM loop is B-loads + WMMA only.
// ---------------------------------------------------------------------------

typedef __attribute__((ext_vector_type(16))) __bf16 v16bf;
typedef __attribute__((ext_vector_type(8)))  float  v8f;

#define BB 64
#define II 32
#define TT 1024
#define HH 512
#define KK 8
#define NG1 20            // phase-1 column groups: i,f,c,o, i_x[8], c_x[8]
#define KT  16            // 512/32 k-tiles
#define MT1 4             // 64/16 row tiles
#define NCT1 640          // 20*512/16 column tiles
#define MT3 36            // 576/16
#define NCT3 32           // 512/16
#define NWG 64
#define TPB 256

// prep element counts
#define N_U1 5242880      // 640 ct * 16 kt * 32 lanes * 16 slots
#define N_W4 65536        // 128 ct * 1 kt * 32 * 16
#define N_WA 262144       // 32 ct * 16 kt * 32 * 16

// workspace byte offsets (all 256B aligned)
#define OFF_SYNC 0u
#define OFF_HA   256u        // h bf16, A-packed:  4*16*32*16 elems  (64 KB)
#define OFF_C    65792u      // c f32 [B,H]                        (128 KB)
#define OFF_ACT  196864u     // activations f32 [B,20,H]           (2.5 MB)
#define OFF_LA   2818304u    // l_all bf16, A-packed: 36*16*32*16  (576 KB)
#define OFF_V    3408128u    // V f32 [576,512]                    (1.1 MB)
#define OFF_U1   4587776u    // U_cat bf16 B-tiles                 (10 MB)
#define OFF_W4   15073536u   // W_cat bf16 B-tiles (K=32)          (128 KB)
#define OFF_WA   15204608u   // W_a bf16 B-tiles                   (512 KB)

struct Args {
  const float* Y;
  const float* x[KK];
  const float* Wm[4];   // W_i W_f W_c W_o
  const float* Um[4];   // U_i U_f U_c U_o
  const float* bm[4];
  const float* W_i_x; const float* U_i_x; const float* b_i_x;
  const float* W_c_x; const float* U_c_x; const float* b_c_x;
  const float* W_a;   const float* b_a;
  float* out;               // h_T (B*H) then hidden_seq (B*T*H)
  unsigned char* ws;
};

// 16-bit A/B tile striping (ISA 7.12.2): lane = {half(1b), m_or_n(4b)},
// slot s (0..15 within v16bf) -> K index inside the 32-wide k-tile.
__device__ __forceinline__ int a_kk(int slot, int half) {
  int v = slot >> 1, p = slot & 1;
  return ((v >> 2) << 4) + ((v & 3) << 1) + (half << 3) + p;
}
__device__ __forceinline__ void a_inv(int kk, int& slot, int& half) {
  int grp = kk >> 4; int r = kk & 15; half = r >> 3; int r2 = r & 7;
  slot = (((grp << 2) + (r2 >> 1)) << 1) + (r2 & 1);
}
__device__ __forceinline__ float sigm(float x) { return 1.f / (1.f + __expf(-x)); }

__device__ __forceinline__ void grid_sync(unsigned* cnt, volatile unsigned* gen,
                                          unsigned nwg) {
  __threadfence();
  __syncthreads();
  if (threadIdx.x == 0) {
    unsigned g0 = *gen;
    if (atomicAdd(cnt, 1u) == nwg - 1u) {
      *cnt = 0u;
      __threadfence();
      atomicAdd((unsigned*)gen, 1u);
    } else {
      while (*gen == g0) { __builtin_amdgcn_s_sleep(1); }
    }
  }
  __syncthreads();
  __threadfence();
}

// ---------------------------------------------------------------------------
__global__ __launch_bounds__(TPB) void init_kernel(Args a) {
  int i = blockIdx.x * blockDim.x + threadIdx.x;
  if (i < 2)       ((unsigned*)(a.ws + OFF_SYNC))[i] = 0u;
  if (i < BB * HH) ((float*)(a.ws + OFF_C))[i] = 0.f;
  if (i < BB * HH) ((__bf16*)(a.ws + OFF_HA))[i] = (__bf16)0.f;
}

// Convert all recurrent weights to bf16 in WMMA B-tile-packed order.
__global__ __launch_bounds__(TPB) void prep_weights(Args a) {
  __bf16* U1 = (__bf16*)(a.ws + OFF_U1);
  __bf16* W4 = (__bf16*)(a.ws + OFF_W4);
  __bf16* WA = (__bf16*)(a.ws + OFF_WA);
  const int ntot = N_U1 + N_W4 + N_WA;
  for (int d = blockIdx.x * blockDim.x + threadIdx.x; d < ntot;
       d += gridDim.x * blockDim.x) {
    if (d < N_U1) {
      int slot = d & 15, lane = (d >> 4) & 31, kt = (d >> 9) & 15, ct = d >> 13;
      int kk = a_kk(slot, lane >> 4);
      int k = kt * 32 + kk;
      int ngl = ct * 16 + (lane & 15);
      int g = ngl >> 9, col = ngl & 511;
      float v;
      if (g < 4)       v = a.Um[g][(size_t)k * HH + col];
      else if (g < 12) v = a.U_i_x[(((size_t)(g - 4)) * HH + k) * HH + col];
      else             v = a.U_c_x[(((size_t)(g - 12)) * HH + k) * HH + col];
      U1[d] = (__bf16)v;
    } else if (d < N_U1 + N_W4) {
      int d2 = d - N_U1;
      int slot = d2 & 15, lane = (d2 >> 4) & 31, ct = d2 >> 9;  // 0..127, 1 k-tile
      int kk = a_kk(slot, lane >> 4);                            // K = 0..31 (I=32)
      int ngl = ct * 16 + (lane & 15);
      int g = ngl >> 9, col = ngl & 511;
      W4[d2] = (__bf16)a.Wm[g][(size_t)kk * HH + col];
    } else {
      int d3 = d - N_U1 - N_W4;
      int slot = d3 & 15, lane = (d3 >> 4) & 31, kt = (d3 >> 9) & 15, ct = d3 >> 13;
      int kk = a_kk(slot, lane >> 4);
      int k = kt * 32 + kk;
      int col = ct * 16 + (lane & 15);
      WA[d3] = (__bf16)a.W_a[(size_t)k * HH + col];
    }
  }
}

// ---------------------------------------------------------------------------
__global__ __launch_bounds__(TPB) void lstm_persistent(Args A) {
  unsigned* cnt = (unsigned*)(A.ws + OFF_SYNC);
  volatile unsigned* gen = ((volatile unsigned*)(A.ws + OFF_SYNC)) + 1;
  const unsigned nwg = gridDim.x;

  const __bf16* U1 = (const __bf16*)(A.ws + OFF_U1);
  const __bf16* W4 = (const __bf16*)(A.ws + OFF_W4);
  const __bf16* WA = (const __bf16*)(A.ws + OFF_WA);
  __bf16* HA  = (__bf16*)(A.ws + OFF_HA);
  __bf16* LA  = (__bf16*)(A.ws + OFF_LA);
  float*  Cst = (float*)(A.ws + OFF_C);
  float*  ACT = (float*)(A.ws + OFF_ACT);
  float*  V   = (float*)(A.ws + OFF_V);

  const int lane  = threadIdx.x & 31;
  const int half  = lane >> 4;
  const int nl    = lane & 15;
  const int gtid  = blockIdx.x * blockDim.x + threadIdx.x;
  const int gsz   = gridDim.x * blockDim.x;
  const int gwave = gtid >> 5;
  const int nwaves = gsz >> 5;

  // Phase-1 wave mapping: fixed row tile, stride over column tiles.
  const int tm1    = gwave & (MT1 - 1);
  const int c01    = gwave >> 2;
  const int cstep1 = nwaves >> 2;            // 128 with the default launch
  // Phase-3 wave mapping: fixed l_all row tile.
  const int tm3    = gwave % MT3;
  const int c03    = gwave / MT3;
  const int cstep3 = nwaves / MT3;           // 14 with the default launch

  for (int t = 0; t < TT; ++t) {
    // ---- Phase 1: z = h @ U_cat (+ input terms), activations -> ACT ----
    {
      // Register-cache this wave's A slab (h row tile, all 16 k-tiles).
      v16bf areg[KT];
      #pragma unroll
      for (int kt = 0; kt < KT; ++kt)
        areg[kt] = *(const v16bf*)(HA + (((size_t)tm1 * KT + kt) * 32 + lane) * 16);
      // Y_t A tile (used only on the first column-tile iteration, ct < 128).
      v16bf ay;
      #pragma unroll
      for (int s = 0; s < 16; ++s) {
        int kk = a_kk(s, half);
        ay[s] = (__bf16)A.Y[(((size_t)(tm1 * 16 + nl)) * II + kk) * TT + t];
      }
      for (int ct = c01; ct < NCT1; ct += cstep1) {
        const __bf16* bbase = U1 + ((size_t)ct * KT) * 32 * 16;
        __builtin_prefetch(bbase + ((size_t)cstep1 * KT) * 32 * 16, 0, 1);
        v8f acc = {};
        #pragma unroll
        for (int kt = 0; kt < KT; ++kt) {
          v16bf bv = *(const v16bf*)(bbase + ((size_t)kt * 32 + lane) * 16);
          acc = __builtin_amdgcn_wmma_f32_16x16x32_bf16(false, areg[kt], false, bv,
                                                        (short)0, acc, false, false);
        }
        const int g = ct >> 5;
        if (g < 4) {                     // main gates: + Y_t @ W_g (K = I = 32)
          v16bf bw = *(const v16bf*)(W4 + ((size_t)ct * 32 + lane) * 16);
          acc = __builtin_amdgcn_wmma_f32_16x16x32_bf16(false, ay, false, bw,
                                                        (short)0, acc, false, false);
        }
        #pragma unroll
        for (int r = 0; r < 8; ++r) {
          const int brow = tm1 * 16 + r + 8 * half;  // C/D layout: M = r (+8 hi half)
          const int ngl  = ct * 16 + nl;
          const int col  = ngl & 511;
          float z = acc[r];
          if (g < 4) {
            z += A.bm[g][col];
          } else if (g < 12) {                       // i_x pathways (x1 for j=0, x2 else)
            const int jj = g - 4;
            const float* xs = (jj == 0) ? A.x[0] : A.x[1];
            float dot = 0.f;
            #pragma unroll
            for (int i = 0; i < 3; ++i)
              dot += xs[(((size_t)brow) * 3 + i) * TT + t] *
                     A.W_i_x[(((size_t)jj) * 3 + i) * HH + col];
            z += dot + A.b_i_x[jj * HH + col];
          } else {                                   // c_x pathways (x_{j+1})
            const int jj = g - 12;
            const float* xs = A.x[jj];
            float dot = 0.f;
            #pragma unroll
            for (int i = 0; i < 3; ++i)
              dot += xs[(((size_t)brow) * 3 + i) * TT + t] *
                     A.W_c_x[(((size_t)jj) * 3 + i) * HH + col];
            z += dot + A.b_c_x[jj * HH + col];
          }
          const bool use_tanh = (g == 2) || (g >= 12);
          ACT[(((size_t)brow) * NG1 + g) * HH + col] = use_tanh ? tanhf(z) : sigm(z);
        }
      }
    }
    grid_sync(cnt, gen, nwg);

    // ---- Phase 2: l_all = gate products -> LA (bf16, A-packed, rows b*9+j) ----
    for (int idx = gtid; idx < BB * 9 * HH; idx += gsz) {
      const int hcol = idx & (HH - 1);
      const int m    = idx >> 9;        // b*9 + j
      const int j    = m % 9;
      const int b    = m / 9;
      const int gi = (j == 0) ? 0 : 4 + (j - 1);
      const int gc = (j == 0) ? 2 : 12 + (j - 1);
      float l = ACT[(((size_t)b) * NG1 + gi) * HH + hcol] *
                ACT[(((size_t)b) * NG1 + gc) * HH + hcol];
      int slot, hf; a_inv(hcol & 31, slot, hf);
      LA[((((size_t)(m >> 4)) * KT + (hcol >> 5)) * 32 + (hf * 16 + (m & 15))) * 16
         + slot] = (__bf16)l;
    }
    grid_sync(cnt, gen, nwg);

    // ---- Phase 3: V = l_all @ W_a  (M=576, N=512, K=512) ----
    if (c03 < cstep3) {
      v16bf areg[KT];
      #pragma unroll
      for (int kt = 0; kt < KT; ++kt)
        areg[kt] = *(const v16bf*)(LA + (((size_t)tm3 * KT + kt) * 32 + lane) * 16);
      for (int ct = c03; ct < NCT3; ct += cstep3) {
        v8f acc = {};
        #pragma unroll
        for (int kt = 0; kt < KT; ++kt) {
          v16bf bv = *(const v16bf*)(WA + (((size_t)ct * KT + kt) * 32 + lane) * 16);
          acc = __builtin_amdgcn_wmma_f32_16x16x32_bf16(false, areg[kt], false, bv,
                                                        (short)0, acc, false, false);
        }
        #pragma unroll
        for (int r = 0; r < 8; ++r)
          V[((size_t)(tm3 * 16 + r + 8 * half)) * HH + ct * 16 + nl] = acc[r];
      }
    }
    grid_sync(cnt, gen, nwg);

    // ---- Phase 4: softmax over 9 pathways, c/h update, outputs ----
    for (int idx = gtid; idx < BB * HH; idx += gsz) {
      const int hcol = idx & (HH - 1);
      const int b    = idx >> 9;
      const float cv = Cst[idx];
      const float f  = ACT[(((size_t)b) * NG1 + 1) * HH + hcol];
      const float o  = ACT[(((size_t)b) * NG1 + 3) * HH + hcol];
      const float ba = A.b_a[hcol];
      float u[9], l[9], umax = -1e30f;
      #pragma unroll
      for (int j = 0; j < 9; ++j) {
        const int gi = (j == 0) ? 0 : 4 + (j - 1);
        const int gc = (j == 0) ? 2 : 12 + (j - 1);
        l[j] = ACT[(((size_t)b) * NG1 + gi) * HH + hcol] *
               ACT[(((size_t)b) * NG1 + gc) * HH + hcol];
        const float v = V[(((size_t)b) * 9 + j) * HH + hcol];
        u[j] = tanhf(v * cv + ba);
        umax = fmaxf(umax, u[j]);
      }
      float se = 0.f, L = 0.f;
      #pragma unroll
      for (int j = 0; j < 9; ++j) {
        const float e = __expf(u[j] - umax);
        se += e; L += e * l[j];
      }
      const float cn = f * cv + L / se;
      const float h  = o * tanhf(cn);
      Cst[idx] = cn;
      A.out[(size_t)BB * HH + (((size_t)b) * TT + t) * HH + hcol] = h;  // hidden_seq
      if (t == TT - 1) A.out[idx] = h;                                   // h_T
      int slot, hf; a_inv(hcol & 31, slot, hf);
      HA[((((size_t)(b >> 4)) * KT + (hcol >> 5)) * 32 + (hf * 16 + (b & 15))) * 16
         + slot] = (__bf16)h;
    }
    grid_sync(cnt, gen, nwg);
  }
}

// ---------------------------------------------------------------------------
extern "C" void kernel_launch(void* const* d_in, const int* in_sizes, int n_in,
                              void* d_out, int out_size, void* d_ws, size_t ws_size,
                              hipStream_t stream) {
  (void)in_sizes; (void)n_in; (void)out_size; (void)ws_size;
  Args a;
  a.Y = (const float*)d_in[0];
  for (int j = 0; j < KK; ++j) a.x[j] = (const float*)d_in[1 + j];
  const int wbase[4] = {9, 12, 15, 18};          // (W,U,b) for i,f,c,o
  for (int g = 0; g < 4; ++g) {
    a.Wm[g] = (const float*)d_in[wbase[g]];
    a.Um[g] = (const float*)d_in[wbase[g] + 1];
    a.bm[g] = (const float*)d_in[wbase[g] + 2];
  }
  a.W_i_x = (const float*)d_in[21]; a.U_i_x = (const float*)d_in[22];
  a.b_i_x = (const float*)d_in[23];
  a.W_c_x = (const float*)d_in[24]; a.U_c_x = (const float*)d_in[25];
  a.b_c_x = (const float*)d_in[26];
  a.W_a = (const float*)d_in[27];   a.b_a = (const float*)d_in[28];
  a.out = (float*)d_out;
  a.ws  = (unsigned char*)d_ws;

  init_kernel<<<(BB * HH + TPB - 1) / TPB, TPB, 0, stream>>>(a);
  const int ntot = N_U1 + N_W4 + N_WA;
  prep_weights<<<(ntot + TPB - 1) / TPB, TPB, 0, stream>>>(a);
  lstm_persistent<<<NWG, TPB, 0, stream>>>(a);
}
